// MemoryEfficientAttention_55722905699066
// MI455X (gfx1250) — matrix operand order repaired and checked by reading, not compiled, not run on hardware
//
#include <hip/hip_runtime.h>
#include <hip/hip_bf16.h>
#include <math.h>

// ---------------------------------------------------------------------------
// MemoryEfficientAttention on MI455X (gfx1250), wave32, bf16 WMMA pipeline.
//   D_MODEL=1024, HEADS=16, HEAD_DIM=64, B=4, S=2048, M = B*S = 8192
// Register-blocked: GEMM waves do 64x32 output (8 wmma / 6 loads per K-step),
// attention waves do 2 query tiles (16 wmma / 8 loads per 32-key chunk).
// ---------------------------------------------------------------------------

#define NHEADS   16
#define HDIM     64
#define SEQ      2048
#define BATCH    4
#define DMODEL   1024
#define MROWS    (BATCH * SEQ)      // 8192
#define ATT_SCALE 0.125f            // 64^-0.5

typedef __attribute__((ext_vector_type(16))) __bf16 v16bf;
typedef __attribute__((ext_vector_type(8)))  float  v8f;

__device__ __forceinline__ v8f wmma_bf16(v16bf a, v16bf b, v8f c) {
  // D = A(16x32) * B(32x16) + C, f32 accumulate
  return __builtin_amdgcn_wmma_f32_16x16x32_bf16(
      /*neg_a=*/false, a, /*neg_b=*/false, b,
      /*c_mod=*/(short)0, c, /*reuse_a=*/false, /*reuse_b=*/false);
}

// ---------------------------------------------------------------------------
// Fragment layouts (ISA 7.12.2, wave32):
//  A 16x32 bf16 : lane l, elem i -> m = l&15, k = (i>>3)*16 + (l>>4)*8 + (i&7)
//  B 32x16 bf16 : lane l, elem i -> n = l&15, k = (i>>3)*16 + (l>>4)*8 + (i&7)
//  C 16x16 f32  : lane l, reg  r -> n = l&15, m = r + 8*(l>>4)
// Packed operand buffers store [tile][lane(32)][elem(16)] contiguously so a
// wave loads one fragment as a single 32B-per-lane vector load.
// ---------------------------------------------------------------------------

// X (f32 [8192][1024]) -> XA packed A-layout [Mt=512][Kt=32][32][16] bf16
__global__ void pack_x(const float* __restrict__ X, __bf16* __restrict__ XA) {
  int idx  = blockIdx.x * blockDim.x + threadIdx.x;   // 512*32*32 lanes
  int lane = idx & 31;
  int kt   = (idx >> 5) & 31;
  int mt   = idx >> 10;
  int m     = mt * 16 + (lane & 15);
  int kbase = kt * 32 + ((lane >> 4) << 3);
  const float* src = X + (size_t)m * DMODEL;
  v16bf v;
#pragma unroll
  for (int j = 0; j < 8; ++j) v[j]     = (__bf16)src[kbase + j];
#pragma unroll
  for (int j = 0; j < 8; ++j) v[8 + j] = (__bf16)src[kbase + 16 + j];
  *(v16bf*)(XA + (size_t)idx * 16) = v;
}

// Wq/Wk/Wv/Wo (f32 [1024][1024]) -> WB packed B-layout [w=4][Nt=64][Kt=32][32][16]
__global__ void pack_w(const float* __restrict__ Wq, const float* __restrict__ Wk,
                       const float* __restrict__ Wv, const float* __restrict__ Wo,
                       __bf16* __restrict__ WB) {
  int idx  = blockIdx.x * blockDim.x + threadIdx.x;   // 4*64*32*32 lanes
  int lane = idx & 31;
  int kt   = (idx >> 5) & 31;
  int nt   = (idx >> 10) & 63;
  int w    = idx >> 16;
  const float* W = (w == 0) ? Wq : (w == 1) ? Wk : (w == 2) ? Wv : Wo;
  int n  = nt * 16 + (lane & 15);
  int k0 = kt * 32 + ((lane >> 4) << 3);
  v16bf v;
#pragma unroll
  for (int i = 0; i < 16; ++i) v[i] = (__bf16)W[(size_t)(k0 + i + ((i >> 3) << 3)) * DMODEL + n];
  *(v16bf*)(WB + (size_t)idx * 16) = v;
}

// Scatter one projected element straight into the attention operand layouts.
__device__ __forceinline__ void scatter_qkv(int z, float val, int row, int col,
                                            __bf16* __restrict__ QB,
                                            __bf16* __restrict__ KA,
                                            __bf16* __restrict__ VB) {
  int bb = row >> 11, s = row & 2047;
  int h  = col >> 6,  d = col & 63;
  int bh = bb * 16 + h;
  if (z == 0) {
    val *= ATT_SCALE;
    int lane2 = (s & 15) + 16 * ((d & 15) >> 3);              // B: n=s%16, k=d
    int i2    = (((d & 31) >> 4) << 3) + (d & 7);
    size_t off = ((((size_t)bh * 128 + (s >> 4)) * 2 + (d >> 5)) * 32 + lane2) * 16 + i2;
    QB[off] = (__bf16)val;
  } else if (z == 1) {
    int kk = d & 31;                                          // A: m=s%16, k=d
    int half2 = (kk & 15) >> 3;
    int i2    = ((kk >> 4) << 3) + (kk & 7);
    int lane2 = (s & 15) + 16 * half2;
    size_t off = ((((size_t)bh * 128 + (s >> 4)) * 2 + (d >> 5)) * 32 + lane2) * 16 + i2;
    KA[off] = (__bf16)val;
  } else {
    int kk = s & 31;                                          // B: n=d%16, k=s
    int lane2 = (d & 15) + 16 * ((kk & 15) >> 3);
    int i2    = ((kk >> 4) << 3) + (kk & 7);
    size_t off = ((((size_t)bh * 4 + (d >> 4)) * 64 + (s >> 5)) * 32 + lane2) * 16 + i2;
    VB[off] = (__bf16)val;
  }
}

// ---------------------------------------------------------------------------
// QKV projection, register-blocked: one wave -> 64x32 output (4 M x 2 N tiles).
// Per K-step: 4 A-frag + 2 B-frag loads feed 8 WMMAs.
//   z=0: Q*scale -> QB  (B-operand of S^T = K*Q^T)  per (b,h): [qt=128][ktd=2][512]
//   z=1: K       -> KA  (A-operand of S^T)          per (b,h): [keyt=128][ktd=2][512]
//   z=2: V       -> VB  (B-operand of P*V)          per (b,h): [dt=4][kc=64][512]
// ---------------------------------------------------------------------------
__global__ void gemm_qkv(const __bf16* __restrict__ XA, const __bf16* __restrict__ WB,
                         const float* __restrict__ bq, const float* __restrict__ bk,
                         const float* __restrict__ bv,
                         __bf16* __restrict__ QB, __bf16* __restrict__ KA,
                         __bf16* __restrict__ VB) {
  int mt0 = blockIdx.x * 4;       // 128 blocks -> 512 M-tiles
  int nt0 = blockIdx.y * 2;       // 32 blocks  -> 64 N-tiles
  int z   = blockIdx.z;           // 0:Q 1:K 2:V
  int lane = threadIdx.x;
  const __bf16* xa = XA + (size_t)mt0 * 32 * 512 + lane * 16;
  const __bf16* wb = WB + ((size_t)z * 64 + nt0) * 32 * 512 + lane * 16;

  v8f acc[4][2] = {};
#pragma unroll 2
  for (int kt = 0; kt < 32; ++kt) {
    v16bf a[4], b[2];
#pragma unroll
    for (int i = 0; i < 4; ++i)
      a[i] = *(const v16bf*)(xa + ((size_t)i * 32 + kt) * 512);
#pragma unroll
    for (int j = 0; j < 2; ++j)
      b[j] = *(const v16bf*)(wb + ((size_t)j * 32 + kt) * 512);
#pragma unroll
    for (int i = 0; i < 4; ++i)
#pragma unroll
      for (int j = 0; j < 2; ++j)
        acc[i][j] = wmma_bf16(a[i], b[j], acc[i][j]);
  }

  const float* bias = (z == 0) ? bq : (z == 1) ? bk : bv;
  int half = lane >> 4;
#pragma unroll
  for (int j = 0; j < 2; ++j) {
    int col = (nt0 + j) * 16 + (lane & 15);
    float badd = bias[col];
#pragma unroll
    for (int i = 0; i < 4; ++i) {
#pragma unroll
      for (int r = 0; r < 8; ++r) {
        int row = (mt0 + i) * 16 + r + 8 * half;
        scatter_qkv(z, acc[i][j][r] + badd, row, col, QB, KA, VB);
      }
    }
  }
}

// ---------------------------------------------------------------------------
// Flash attention: one wave per (b, h, 2 query tiles); 64 chunks of 32 keys.
// S^T = K*Q^T in C-layout (lane<->query col, regs<->key row) feeds P directly
// as the A-fragment of P*V -- zero cross-lane transposes needed.
// K-fragments reused across both query tiles; V-fragments across both ctx sets.
// ---------------------------------------------------------------------------
__global__ void attn(const __bf16* __restrict__ QB, const __bf16* __restrict__ KA,
                     const __bf16* __restrict__ VB, __bf16* __restrict__ CA) {
  int qt0 = blockIdx.x * 2;       // 64 blocks -> 128 q-tiles
  int h   = blockIdx.y;           // 16
  int b   = blockIdx.z;           // 4
  int lane = threadIdx.x;
  int half = lane >> 4;
  int bh = b * NHEADS + h;

  const __bf16* qbp = QB + (((size_t)bh * 128 + qt0) * 2) * 512 + lane * 16;
  v16bf qb[2][2];
#pragma unroll
  for (int qi = 0; qi < 2; ++qi) {
    qb[qi][0] = *(const v16bf*)(qbp + (size_t)(qi * 2 + 0) * 512);
    qb[qi][1] = *(const v16bf*)(qbp + (size_t)(qi * 2 + 1) * 512);
  }
  const __bf16* kap = KA + (size_t)bh * 128 * 2 * 512 + lane * 16;
  const __bf16* vbp = VB + (size_t)bh * 4 * 64 * 512 + lane * 16;

  v8f cc[2][4] = {};
  float mrun[2] = {-1.0e30f, -1.0e30f};
  float ssum[2] = {0.0f, 0.0f};
  v16bf pa[2];

  for (int kc = 0; kc < 64; ++kc) {
    const __bf16* kp = kap + (size_t)kc * 2048;          // 2 key tiles x 2 d-steps
    v16bf a00 = *(const v16bf*)(kp);
    v16bf a01 = *(const v16bf*)(kp + 512);
    v16bf a10 = *(const v16bf*)(kp + 1024);
    v16bf a11 = *(const v16bf*)(kp + 1536);

#pragma unroll
    for (int qi = 0; qi < 2; ++qi) {
      v8f s0 = {}, s1 = {};
      s0 = wmma_bf16(a00, qb[qi][0], s0);
      s0 = wmma_bf16(a01, qb[qi][1], s0);
      s1 = wmma_bf16(a10, qb[qi][0], s1);
      s1 = wmma_bf16(a11, qb[qi][1], s1);

      // online softmax over key dim (= rows of S^T): in-reg + shfl_xor(16)
      float cm = s0[0];
#pragma unroll
      for (int r = 1; r < 8; ++r) cm = fmaxf(cm, s0[r]);
#pragma unroll
      for (int r = 0; r < 8; ++r) cm = fmaxf(cm, s1[r]);
      cm = fmaxf(cm, __shfl_xor(cm, 16));
      float nm = fmaxf(mrun[qi], cm);
      float alpha = __expf(mrun[qi] - nm);

      float p[16], ps = 0.0f;
#pragma unroll
      for (int r = 0; r < 8; ++r) { p[r]     = __expf(s0[r] - nm); ps += p[r]; }
#pragma unroll
      for (int r = 0; r < 8; ++r) { p[8 + r] = __expf(s1[r] - nm); ps += p[8 + r]; }
      ps += __shfl_xor(ps, 16);
      ssum[qi] = ssum[qi] * alpha + ps;
      mrun[qi] = nm;

      // rescale running ctx: reg r holds query m = r + 8*half -> fetch its alpha
#pragma unroll
      for (int r = 0; r < 8; ++r) {
        float f = __shfl(alpha, r + 8 * half);
        cc[qi][0][r] *= f; cc[qi][1][r] *= f; cc[qi][2][r] *= f; cc[qi][3][r] *= f;
      }

      // P already matches the A-fragment layout for K=32 (regs 0..7 of both tiles)
#pragma unroll
      for (int j = 0; j < 16; ++j) pa[qi][j] = (__bf16)p[j];
    }

    const __bf16* vb = vbp + (size_t)kc * 512;
#pragma unroll
    for (int t = 0; t < 4; ++t) {
      v16bf bfrag = *(const v16bf*)(vb + (size_t)t * 64 * 512);
      cc[0][t] = wmma_bf16(pa[0], bfrag, cc[0][t]);
      cc[1][t] = wmma_bf16(pa[1], bfrag, cc[1][t]);
    }
  }

#pragma unroll
  for (int qi = 0; qi < 2; ++qi) {
    float inv = 1.0f / ssum[qi];
#pragma unroll
    for (int r = 0; r < 8; ++r) {
      float g = __shfl(inv, r + 8 * half);
      cc[qi][0][r] *= g; cc[qi][1][r] *= g; cc[qi][2][r] *= g; cc[qi][3][r] *= g;
    }
    // scatter ctx into CA packed A-layout [Mt=512][Kt=32][32][16]
#pragma unroll
    for (int t = 0; t < 4; ++t) {
#pragma unroll
      for (int r = 0; r < 8; ++r) {
        int row = b * SEQ + (qt0 + qi) * 16 + r + 8 * half;
        int col = h * HDIM + t * 16 + (lane & 15);
        int kk = col & 31;
        int half2 = (kk & 15) >> 3;
        int i2 = ((kk >> 4) << 3) + (kk & 7);
        int lane2 = (row & 15) + 16 * half2;
        size_t off = (((size_t)(row >> 4) * 32 + (col >> 5)) * 32 + lane2) * 16 + i2;
        CA[off] = (__bf16)cc[qi][t][r];
      }
    }
  }
}

// Output projection, register-blocked 64x32: ctx(CA) x Wo + bo -> f32 out
__global__ void gemm_out(const __bf16* __restrict__ CA, const __bf16* __restrict__ WB,
                         const float* __restrict__ bo, float* __restrict__ out) {
  int mt0 = blockIdx.x * 4;       // 128 blocks -> 512 M-tiles
  int nt0 = blockIdx.y * 2;       // 32 blocks  -> 64 N-tiles
  int lane = threadIdx.x;
  const __bf16* ca = CA + (size_t)mt0 * 32 * 512 + lane * 16;
  const __bf16* wb = WB + ((size_t)3 * 64 + nt0) * 32 * 512 + lane * 16;  // w=3 -> Wo

  v8f acc[4][2] = {};
#pragma unroll 2
  for (int kt = 0; kt < 32; ++kt) {
    v16bf a[4], b[2];
#pragma unroll
    for (int i = 0; i < 4; ++i)
      a[i] = *(const v16bf*)(ca + ((size_t)i * 32 + kt) * 512);
#pragma unroll
    for (int j = 0; j < 2; ++j)
      b[j] = *(const v16bf*)(wb + ((size_t)j * 32 + kt) * 512);
#pragma unroll
    for (int i = 0; i < 4; ++i)
#pragma unroll
      for (int j = 0; j < 2; ++j)
        acc[i][j] = wmma_bf16(a[i], b[j], acc[i][j]);
  }

  int half = lane >> 4;
#pragma unroll
  for (int j = 0; j < 2; ++j) {
    int col = (nt0 + j) * 16 + (lane & 15);
    float bias = bo[col];
#pragma unroll
    for (int i = 0; i < 4; ++i) {
#pragma unroll
      for (int r = 0; r < 8; ++r) {
        int row = (mt0 + i) * 16 + r + 8 * half;
        out[(size_t)row * DMODEL + col] = acc[i][j][r] + bias;
      }
    }
  }
}

// ---------------------------------------------------------------------------
extern "C" void kernel_launch(void* const* d_in, const int* in_sizes, int n_in,
                              void* d_out, int out_size, void* d_ws, size_t ws_size,
                              hipStream_t stream) {
  const float* X  = (const float*)d_in[0];
  const float* Wq = (const float*)d_in[1];
  const float* bq = (const float*)d_in[2];
  const float* Wk = (const float*)d_in[3];
  const float* bk = (const float*)d_in[4];
  const float* Wv = (const float*)d_in[5];
  const float* bv = (const float*)d_in[6];
  const float* Wo = (const float*)d_in[7];
  const float* bo = (const float*)d_in[8];
  float* out = (float*)d_out;

  char* ws = (char*)d_ws;
  // 88 MB of packed bf16 operands
  __bf16* XA = (__bf16*)(ws);                          // 16 MB
  __bf16* WB = (__bf16*)(ws + (16ull << 20));          //  8 MB (q,k,v,o)
  __bf16* QB = (__bf16*)(ws + (24ull << 20));          // 16 MB
  __bf16* KA = (__bf16*)(ws + (40ull << 20));          // 16 MB
  __bf16* VB = (__bf16*)(ws + (56ull << 20));          // 16 MB
  __bf16* CA = (__bf16*)(ws + (72ull << 20));          // 16 MB

  pack_x<<<2048, 256, 0, stream>>>(X, XA);                       // 512*32*32 lanes
  pack_w<<<1024, 256, 0, stream>>>(Wq, Wk, Wv, Wo, WB);          // 4*64*32*32 lanes
  gemm_qkv<<<dim3(128, 32, 3), 32, 0, stream>>>(XA, WB, bq, bk, bv, QB, KA, VB);
  attn<<<dim3(64, NHEADS, BATCH), 32, 0, stream>>>(QB, KA, VB, CA);
  gemm_out<<<dim3(128, 32), 32, 0, stream>>>(CA, WB, bo, out);
}
